// MultiHeadAttention_35502199668770
// MI455X (gfx1250) — matrix operand order repaired and checked by reading, not compiled, not run on hardware
//
#include <hip/hip_runtime.h>
#include <hip/hip_bf16.h>

// ---------------- CDNA5 WMMA types ----------------
typedef __attribute__((ext_vector_type(16))) _Float16 v16h;
typedef __attribute__((ext_vector_type(8)))  float    v8f;
typedef __attribute__((ext_vector_type(4)))  _Float16 h4;

union FragU { uint4 q[2]; v16h h; };

__device__ __forceinline__ v8f wmma16(v16h a, v16h b, v8f c) {
  // D = A(16x32 f16) * B(32x16 f16) + C(16x16 f32)
  return __builtin_amdgcn_wmma_f32_16x16x32_f16(
      /*neg_a=*/false, a, /*neg_b=*/false, b,
      /*c_mod=*/(short)0, c, /*reuse_a=*/false, /*reuse_b=*/false);
}

// A fragment: lane row fixed; K chunks [half*8 .. +7] and [16+half*8 .. +7]
__device__ __forceinline__ v16h load_frag_a(const _Float16* rowPtr, int half) {
  FragU f;
  f.q[0] = *(const uint4*)(rowPtr + half * 8);
  f.q[1] = *(const uint4*)(rowPtr + 16 + half * 8);
  return f.h;
}
// B fragment: lane col fixed; K chunks [halfsel*8 .. +7] and [16+halfsel*8 .. +7]
__device__ __forceinline__ v16h load_frag_b(const _Float16* rowPtr, int halfsel) {
  FragU f;
  f.q[0] = *(const uint4*)(rowPtr + halfsel * 8);
  f.q[1] = *(const uint4*)(rowPtr + 16 + halfsel * 8);
  return f.h;
}

// Async global->LDS copy, 16B per lane, tracked by ASYNCcnt.
// Generic pointer to a __shared__ object carries the workgroup-relative LDS byte
// offset in its low 32 bits (flat LDS aperture mapping), which is what the
// instruction's LDS-address VGPR expects.
__device__ __forceinline__ void async_copy_b128(const _Float16* gp, _Float16* lp) {
  unsigned loff = (unsigned)(unsigned long long)lp;
  asm volatile("global_load_async_to_lds_b128 %0, %1, off"
               :: "v"(loff), "v"((unsigned long long)gp)
               : "memory");
}
__device__ __forceinline__ void wait_async0() {
  asm volatile("s_wait_asynccnt 0x0" ::: "memory");
}

#define Bsz 4
#define Ssz 2048
#define Hh  16
#define Dd  1024

// ------------- weight fp32 [K][N] -> f16 [N][K] -------------
__global__ __launch_bounds__(256) void mha_wtrans(const float* __restrict__ W,
                                                  _Float16* __restrict__ WT) {
  int idx = blockIdx.x * 256 + threadIdx.x;          // 1M elements
  int k = idx >> 10, n = idx & 1023;
  WT[(size_t)n * 1024 + k] = (_Float16)W[(size_t)k * 1024 + n];
}

// ------------- QKV projection: X(fp32)[8192x1024] @ WT^T + b -> head-major f16
// mode 0: Q, scaled 1/8, out [b,h,s,d]; mode 1: K, [b,h,s,d]; mode 2: V, [b,h,d,s]
__global__ __launch_bounds__(256) void mha_gemm_qkv(const float* __restrict__ X,
                                                    const _Float16* __restrict__ WT,
                                                    const float* __restrict__ bias,
                                                    _Float16* __restrict__ out,
                                                    int mode) {
  __shared__ alignas(16) _Float16 As[128][40];
  __shared__ alignas(16) _Float16 Bs[128][40];
  const int tid = threadIdx.x, lane = tid & 31, wid = tid >> 5;
  const int wm = wid & 3, wn = wid >> 2;
  const int half = lane >> 4, l16 = lane & 15;
  const int mBase = blockIdx.y * 128;
  const int nBase = blockIdx.x * 128;

  v8f zero = {};
  v8f acc[2][4];
#pragma unroll
  for (int i = 0; i < 2; ++i)
#pragma unroll
    for (int j = 0; j < 4; ++j) acc[i][j] = zero;

#pragma unroll 4
  for (int kt = 0; kt < 1024; kt += 32) {
    {  // B: 128x32 f16 tile via async global->LDS (WT is [N][K] row-major)
      int tc = tid & 3, tr = tid >> 2;
#pragma unroll
      for (int p = 0; p < 2; ++p) {
        int r = p * 64 + tr;
        async_copy_b128(WT + (size_t)(nBase + r) * 1024 + kt + tc * 8,
                        &Bs[r][tc * 8]);
      }
    }
    {  // A: 128x32 fp32 -> f16 convert through VGPRs
      int tc = tid & 7, tr = tid >> 3;
#pragma unroll
      for (int p = 0; p < 4; ++p) {
        int r = p * 32 + tr;
        float4 v = *(const float4*)(X + (size_t)(mBase + r) * 1024 + kt + tc * 4);
        h4 hv = {(_Float16)v.x, (_Float16)v.y, (_Float16)v.z, (_Float16)v.w};
        *(h4*)&As[r][tc * 4] = hv;
      }
    }
    wait_async0();
    __syncthreads();
#pragma unroll
    for (int mi = 0; mi < 2; ++mi) {
      v16h a = load_frag_a(&As[wm * 32 + mi * 16 + l16][0], half);
#pragma unroll
      for (int ni = 0; ni < 4; ++ni) {
        v16h b = load_frag_b(&Bs[wn * 64 + ni * 16 + l16][0], half);
        acc[mi][ni] = wmma16(a, b, acc[mi][ni]);
      }
    }
    __syncthreads();
  }

  const float scl = (mode == 0) ? 0.125f : 1.0f;
#pragma unroll
  for (int mi = 0; mi < 2; ++mi)
#pragma unroll
    for (int ni = 0; ni < 4; ++ni) {
      int n = nBase + wn * 64 + ni * 16 + l16;
      float bn = bias[n];
      int h = n >> 6, d = n & 63;
#pragma unroll
      for (int r = 0; r < 8; ++r) {
        int m = mBase + wm * 32 + mi * 16 + r + half * 8;
        int b = m >> 11, s = m & (Ssz - 1);
        float val = (acc[mi][ni][r] + bn) * scl;
        size_t off = (mode == 2)
                         ? (((size_t)(b * Hh + h) * 64 + d) * Ssz + s)
                         : (((size_t)(b * Hh + h) * Ssz + s) * 64 + d);
        out[off] = (_Float16)val;
      }
    }
}

// ------------- flash attention: per (b,h), 128 queries/block, 64-key tiles ---
__global__ __launch_bounds__(256) void mha_flash(const _Float16* __restrict__ Qh,
                                                 const _Float16* __restrict__ Kh,
                                                 const _Float16* __restrict__ Vt,
                                                 const float* __restrict__ vmask,
                                                 _Float16* __restrict__ outA) {
  __shared__ alignas(16) _Float16 Ks[64][72];      // [key][d]
  __shared__ alignas(16) _Float16 Vs[64][72];      // [d][key]  (V^T)
  __shared__ alignas(16) _Float16 Ps[8][16][72];   // wave-private P panels
  __shared__ float Ms[64];

  const int tid = threadIdx.x, lane = tid & 31, wid = tid >> 5;
  const int half = lane >> 4, l16 = lane & 15;
  const int bh = blockIdx.y, batch = bh >> 4, hh = bh & 15;
  const int qBase = blockIdx.x * 128;
  const int qRow = qBase + wid * 16 + l16;

  // Q fragments resident for whole kernel (Q pre-scaled by 1/8)
  const _Float16* Qp = Qh + ((size_t)bh * Ssz + qRow) * 64;
  v16h qa0 = load_frag_a(Qp, half);
  v16h qa1 = load_frag_a(Qp + 32, half);

  v8f zero = {};
  v8f oacc[4];
#pragma unroll
  for (int i = 0; i < 4; ++i) oacc[i] = zero;
  float rmax[8], rsum[8];
#pragma unroll
  for (int r = 0; r < 8; ++r) { rmax[r] = -3.0e38f; rsum[r] = 0.0f; }

#pragma unroll 2
  for (int kt = 0; kt < Ssz; kt += 64) {
    __syncthreads();
    {  // K tile 64x64 and V^T tile 64x64 via async global->LDS
      int tc = tid & 7, tr = tid >> 3;
#pragma unroll
      for (int p = 0; p < 2; ++p) {
        int r = p * 32 + tr;
        async_copy_b128(Kh + ((size_t)bh * Ssz + kt + r) * 64 + tc * 8,
                        &Ks[r][tc * 8]);
        async_copy_b128(Vt + ((size_t)bh * 64 + r) * Ssz + kt + tc * 8,
                        &Vs[r][tc * 8]);
      }
    }
    if (tid < 64) Ms[tid] = (vmask[batch * Ssz + kt + tid] - 1.0f) * 1.0e12f;
    wait_async0();
    __syncthreads();

    // ---- S = Q K^T + mask : 16 q-rows x 64 keys per wave ----
    float sv[4][8];
#pragma unroll
    for (int ni = 0; ni < 4; ++ni) {
      const _Float16* kr = &Ks[ni * 16 + l16][0];
      v8f c = zero;
      c = wmma16(qa0, load_frag_b(kr, half), c);
      c = wmma16(qa1, load_frag_b(kr + 32, half), c);
      float madd = Ms[ni * 16 + l16];
#pragma unroll
      for (int r = 0; r < 8; ++r) sv[ni][r] = c[r] + madd;
    }

    // ---- online softmax (row stats wave-local; reduce over 16-lane group) ---
    float nmax[8], corr[8], part[8];
#pragma unroll
    for (int r = 0; r < 8; ++r) {
      nmax[r] = rmax[r];
#pragma unroll
      for (int ni = 0; ni < 4; ++ni) nmax[r] = fmaxf(nmax[r], sv[ni][r]);
      for (int x = 1; x < 16; x <<= 1)
        nmax[r] = fmaxf(nmax[r], __shfl_xor(nmax[r], x, 32));
      corr[r] = __expf(rmax[r] - nmax[r]);
      rmax[r] = nmax[r];
      part[r] = 0.0f;
    }
#pragma unroll
    for (int ni = 0; ni < 4; ++ni) {
      int col = ni * 16 + l16;
#pragma unroll
      for (int r = 0; r < 8; ++r) {
        float p = __expf(sv[ni][r] - nmax[r]);
        part[r] += p;
        Ps[wid][r + half * 8][col] = (_Float16)p;
      }
    }
#pragma unroll
    for (int r = 0; r < 8; ++r) {
      for (int x = 1; x < 16; x <<= 1) part[r] += __shfl_xor(part[r], x, 32);
      rsum[r] = rsum[r] * corr[r] + part[r];
    }
#pragma unroll
    for (int d4 = 0; d4 < 4; ++d4)
#pragma unroll
      for (int r = 0; r < 8; ++r) oacc[d4][r] *= corr[r];
    __syncthreads();

    // ---- O += P V : A = P[16x64], B = V[key][d] read from Vs[d][key] ----
#pragma unroll
    for (int t = 0; t < 2; ++t) {
      v16h pa = load_frag_a(&Ps[wid][l16][t * 32], half);
#pragma unroll
      for (int d4 = 0; d4 < 4; ++d4) {
        v16h vb = load_frag_b(&Vs[d4 * 16 + l16][t * 32], half);
        oacc[d4] = wmma16(pa, vb, oacc[d4]);
      }
    }
  }

  // ---- normalize + store f16 [token][h*64+d] ----
#pragma unroll
  for (int d4 = 0; d4 < 4; ++d4)
#pragma unroll
    for (int r = 0; r < 8; ++r) {
      int row = qBase + wid * 16 + r + half * 8;
      size_t tok = (size_t)batch * Ssz + row;
      int col = hh * 64 + d4 * 16 + l16;
      float v = oacc[d4][r] / rsum[r];
      outA[tok * 1024 + col] = (_Float16)v;
    }
}

// ------------- output projection: attn(f16) @ WoT + bo, * q_mask, fp32 out ---
// Double-buffered LDS with async copy/compute overlap.
__global__ __launch_bounds__(256) void mha_gemm_out(const _Float16* __restrict__ A,
                                                    const _Float16* __restrict__ WT,
                                                    const float* __restrict__ bias,
                                                    const float* __restrict__ qmask,
                                                    float* __restrict__ out) {
  __shared__ alignas(16) _Float16 As[2][128][40];
  __shared__ alignas(16) _Float16 Bs[2][128][40];
  const int tid = threadIdx.x, lane = tid & 31, wid = tid >> 5;
  const int wm = wid & 3, wn = wid >> 2;
  const int half = lane >> 4, l16 = lane & 15;
  const int mBase = blockIdx.y * 128;
  const int nBase = blockIdx.x * 128;

  auto issue = [&](int buf, int kt) {
    int tc = tid & 3, tr = tid >> 2;
#pragma unroll
    for (int p = 0; p < 2; ++p) {
      int r = p * 64 + tr;
      async_copy_b128(A + (size_t)(mBase + r) * 1024 + kt + tc * 8,
                      &As[buf][r][tc * 8]);
      async_copy_b128(WT + (size_t)(nBase + r) * 1024 + kt + tc * 8,
                      &Bs[buf][r][tc * 8]);
    }
  };

  v8f zero = {};
  v8f acc[2][4];
#pragma unroll
  for (int i = 0; i < 2; ++i)
#pragma unroll
    for (int j = 0; j < 4; ++j) acc[i][j] = zero;

  issue(0, 0);
#pragma unroll 4
  for (int it = 0; it < 32; ++it) {
    const int kt = it * 32;
    const int cur = it & 1;
    wait_async0();
    __syncthreads();                 // buf[cur] visible to all waves
    if (kt + 32 < 1024) issue(cur ^ 1, kt + 32);  // overlap next copy
#pragma unroll
    for (int mi = 0; mi < 2; ++mi) {
      v16h a = load_frag_a(&As[cur][wm * 32 + mi * 16 + l16][0], half);
#pragma unroll
      for (int ni = 0; ni < 4; ++ni) {
        v16h b = load_frag_b(&Bs[cur][wn * 64 + ni * 16 + l16][0], half);
        acc[mi][ni] = wmma16(a, b, acc[mi][ni]);
      }
    }
    __syncthreads();                 // all done reading buf[cur] before reuse
  }

#pragma unroll
  for (int mi = 0; mi < 2; ++mi)
#pragma unroll
    for (int ni = 0; ni < 4; ++ni) {
      int n = nBase + wn * 64 + ni * 16 + l16;
      float bn = bias[n];
#pragma unroll
      for (int r = 0; r < 8; ++r) {
        int m = mBase + wm * 32 + mi * 16 + r + half * 8;
        float val = (acc[mi][ni][r] + bn) * qmask[m];
        out[(size_t)m * 1024 + n] = val;
      }
    }
}

// ---------------------------------------------------------------------------
extern "C" void kernel_launch(void* const* d_in, const int* in_sizes, int n_in,
                              void* d_out, int out_size, void* d_ws, size_t ws_size,
                              hipStream_t stream) {
  const float* q     = (const float*)d_in[0];
  const float* k     = (const float*)d_in[1];
  const float* v     = (const float*)d_in[2];
  const float* vmask = (const float*)d_in[3];
  const float* qmask = (const float*)d_in[4];
  const float* Wq    = (const float*)d_in[5];
  const float* bq    = (const float*)d_in[6];
  const float* Wk    = (const float*)d_in[7];
  const float* bk    = (const float*)d_in[8];
  const float* Wv    = (const float*)d_in[9];
  const float* bv    = (const float*)d_in[10];
  const float* Wo    = (const float*)d_in[11];
  const float* bo    = (const float*)d_in[12];

  _Float16* ws = (_Float16*)d_ws;
  const size_t MH = (size_t)1024 * 1024;  // halves per weight matrix
  _Float16* WqT = ws + 0 * MH;
  _Float16* WkT = ws + 1 * MH;
  _Float16* WvT = ws + 2 * MH;
  _Float16* WoT = ws + 3 * MH;
  _Float16* Qh  = ws + 4 * MH;            // [b,h,s,64]  (pre-scaled 1/8)
  _Float16* Kh  = ws + 12 * MH;           // [b,h,s,64]
  _Float16* Vt  = ws + 20 * MH;           // [b,h,64,s]
  _Float16* AO  = ws + 28 * MH;           // [token,1024]

  mha_wtrans<<<4096, 256, 0, stream>>>(Wq, WqT);
  mha_wtrans<<<4096, 256, 0, stream>>>(Wk, WkT);
  mha_wtrans<<<4096, 256, 0, stream>>>(Wv, WvT);
  mha_wtrans<<<4096, 256, 0, stream>>>(Wo, WoT);

  dim3 gg(8, 64);
  mha_gemm_qkv<<<gg, 256, 0, stream>>>(q, WqT, bq, Qh, 0);
  mha_gemm_qkv<<<gg, 256, 0, stream>>>(k, WkT, bk, Kh, 1);
  mha_gemm_qkv<<<gg, 256, 0, stream>>>(v, WvT, bv, Vt, 2);

  mha_flash<<<dim3(16, 64), 256, 0, stream>>>(Qh, Kh, Vt, vmask, AO);

  mha_gemm_out<<<gg, 256, 0, stream>>>(AO, WoT, bo, qmask, (float*)d_out);
}
